// DeepseekV4Compressor_67637144978438
// MI455X (gfx1250) — matrix operand order repaired and checked by reading, not compiled, not run on hardware
//
#include <hip/hip_runtime.h>
#include <hip/hip_bf16.h>
#include <stdint.h>

// ---------------- problem constants ----------------
#define BATCH 4
#define SEQ   4096
#define HID   4096
#define HD    512
#define ROPE  64
#define RATIO 4
#define CHK   (SEQ / RATIO)      // 1024
#define OUTC  (2 * HD)           // 1024 columns of kv/gate
#define EPS   1e-6f

// ---------------- GEMM tiling ----------------
#define MT 128                   // M rows per workgroup
#define NT 128                   // kv columns per workgroup (gate adds another NT)
#define KT 32                    // K per step (matches 16x16x32 bf16 WMMA)
#define LDR 40                   // padded LDS row stride in ushorts (80B, 16B-aligned)

typedef __attribute__((ext_vector_type(16))) __bf16 v16bf;
typedef __attribute__((ext_vector_type(8)))  float  v8f;

union Frag16 { v16bf v; uint4 q[2]; };

__device__ __forceinline__ unsigned short f2bf(float f) {
    unsigned int u = __float_as_uint(f);
    unsigned int r = u + 0x7FFFu + ((u >> 16) & 1u);   // round-to-nearest-even
    return (unsigned short)(r >> 16);
}

// CDNA5 async global->LDS copy (16B per lane), tracked with ASYNCcnt.
__device__ __forceinline__ void async_b128(unsigned lds_off, const void* gaddr) {
    asm volatile("global_load_async_to_lds_b128 %0, %1, off"
                 :: "v"(lds_off), "v"(gaddr) : "memory");
}
__device__ __forceinline__ void wait_async0() {
    asm volatile("s_wait_asynccnt 0" ::: "memory");
}

// ---------------------------------------------------------------------------
// Kernel 0: f32 -> bf16 conversion (bandwidth bound, 8 elems/thread)
// ---------------------------------------------------------------------------
__global__ __launch_bounds__(256)
void cvt_bf16_kernel(const float* __restrict__ in,
                     unsigned short* __restrict__ out, int n8) {
    int i = blockIdx.x * 256 + threadIdx.x;
    if (i >= n8) return;
    const float4* p = (const float4*)in + (size_t)i * 2;
    float4 a = p[0], b = p[1];
    union { uint4 q; unsigned short s[8]; } pk;
    const float* f0 = (const float*)&a;
    const float* f1 = (const float*)&b;
#pragma unroll
    for (int e = 0; e < 4; ++e) pk.s[e] = f2bf(f0[e]);
#pragma unroll
    for (int e = 0; e < 4; ++e) pk.s[4 + e] = f2bf(f1[e]);
    ((uint4*)out)[i] = pk.q;
}

// ---------------------------------------------------------------------------
// Kernel 1: softmax weights from ape (RATIO x 2*HD) -> w8 (8x512), w0 (4x512)
// ---------------------------------------------------------------------------
__global__ void softmax_w_kernel(const float* __restrict__ ape,
                                 float* __restrict__ w8,
                                 float* __restrict__ w0) {
    int d = blockIdx.x * blockDim.x + threadIdx.x;
    if (d >= HD) return;
    float a[8];
#pragma unroll
    for (int r = 0; r < 4; ++r) {
        a[r]     = ape[r * OUTC + d];
        a[4 + r] = ape[r * OUTC + HD + d];
    }
    float m8 = a[0];
#pragma unroll
    for (int i = 1; i < 8; ++i) m8 = fmaxf(m8, a[i]);
    float s8 = 0.f, e8[8];
#pragma unroll
    for (int i = 0; i < 8; ++i) { e8[i] = __expf(a[i] - m8); s8 += e8[i]; }
    float inv8 = 1.f / s8;
#pragma unroll
    for (int i = 0; i < 8; ++i) w8[i * HD + d] = e8[i] * inv8;
    float m4 = fmaxf(fmaxf(a[0], a[1]), fmaxf(a[2], a[3]));
    float s4 = 0.f, e4[4];
#pragma unroll
    for (int i = 0; i < 4; ++i) { e4[i] = __expf(a[i] - m4); s4 += e4[i]; }
    float inv4 = 1.f / s4;
#pragma unroll
    for (int i = 0; i < 4; ++i) w0[i * HD + d] = e4[i] * inv4;
}

// ---------------------------------------------------------------------------
// Kernel 2: fused dual GEMM with sigmoid gating (bf16 inputs, f32 accum).
//   out[m][n] = (x@Wkv^T) * sigmoid(x@Wg^T),  m in [0,16384), n in [0,1024)
// Tiles staged global->LDS via GLOBAL_LOAD_ASYNC_TO_LDS_B128, double-buffered,
// one barrier + one s_wait_asynccnt per K-step; V_WMMA_F32_16X16X32_BF16 core.
// Block: 256 threads = 8 waves arranged 4(M) x 2(N). Wave tile 32x64 (x2 mats).
// ---------------------------------------------------------------------------
__global__ __launch_bounds__(256)
void gemm_gated_kernel(const unsigned short* __restrict__ xbf,
                       const unsigned short* __restrict__ Wkvbf,
                       const unsigned short* __restrict__ Wgbf,
                       float* __restrict__ kvout) {
    __shared__ unsigned short sA[2][MT][LDR];        // 2 x (128 x 32 bf16, padded)
    __shared__ unsigned short sB[2][2 * NT][LDR];    // 2 x (kv rows | gate rows)

    const int t    = threadIdx.x;
    const int m0   = blockIdx.x * MT;
    const int n0   = blockIdx.y * NT;
    const int wave = t >> 5;
    const int lane = t & 31;
    const int wm   = wave & 3;      // M group of 32
    const int wn   = wave >> 2;     // N group of 64

    // ---- loader setup (per-thread fixed addresses) ----
    const int arow = t >> 1;
    const int ahalf = t & 1;                              // 16 bf16 each
    const unsigned short* aRow = xbf + (size_t)(m0 + arow) * HID + ahalf * 16;
    const unsigned short* bRow = (t < NT)
        ? (Wkvbf + (size_t)(n0 + t) * HID)
        : (Wgbf  + (size_t)(n0 + (t - NT)) * HID);

    const unsigned ldsA0 = (unsigned)(size_t)&sA[0][0][0];
    const unsigned ldsB0 = (unsigned)(size_t)&sB[0][0][0];
    const unsigned SZA = MT * LDR * 2;                    // bytes per A buffer
    const unsigned SZB = 2 * NT * LDR * 2;                // bytes per B buffer
    const unsigned laBase = ldsA0 + arow * (LDR * 2) + ahalf * 32;
    const unsigned lbBase = ldsB0 + t * (LDR * 2);

    v8f acc_kv[2][4];
    v8f acc_g[2][4];
#pragma unroll
    for (int i = 0; i < 2; ++i)
#pragma unroll
        for (int j = 0; j < 4; ++j) {
            acc_kv[i][j] = (v8f)(0.0f);
            acc_g[i][j]  = (v8f)(0.0f);
        }

    const int mlo   = lane & 15;
    const int khalf = lane >> 4;

    // issue async copies for one K-tile into buffer `buf`
    auto issue_tile = [&](int k0, int buf) {
        const unsigned la = laBase + buf * SZA;
        const unsigned short* ga = aRow + k0;
        async_b128(la,      ga);          // A: 16 bf16
        async_b128(la + 16, ga + 8);
        const unsigned lb = lbBase + buf * SZB;
        const unsigned short* gb = bRow + k0;
        async_b128(lb,      gb);          // B: 32 bf16 (one W row slice)
        async_b128(lb + 16, gb + 8);
        async_b128(lb + 32, gb + 16);
        async_b128(lb + 48, gb + 24);
    };

    issue_tile(0, 0);
    int cur = 0;

    for (int k0 = 0; k0 < HID; k0 += KT) {
        wait_async0();        // this wave's copies into buf `cur` done
        __syncthreads();      // everyone's copies done; prev buf fully consumed
        if (k0 + KT < HID)
            issue_tile(k0 + KT, cur ^ 1);   // overlap next copy with compute

        // ---- fragments + 16 WMMAs from buffer `cur` ----
        Frag16 afrag[2];
#pragma unroll
        for (int mf = 0; mf < 2; ++mf) {
            const int m  = wm * 32 + mf * 16 + mlo;
            const int kb = khalf * 16;     // elems 0..15 -> K kb..kb+15
            afrag[mf].q[0] = *(const uint4*)&sA[cur][m][kb];
            afrag[mf].q[1] = *(const uint4*)&sA[cur][m][kb + 8];
        }
#pragma unroll
        for (int nf = 0; nf < 4; ++nf) {
            const int n   = wn * 64 + nf * 16 + mlo;
            const int kb2 = khalf * 16;    // 16 consecutive K per half-wave
            Frag16 bkv, bg;
            bkv.q[0] = *(const uint4*)&sB[cur][n][kb2];
            bkv.q[1] = *(const uint4*)&sB[cur][n][kb2 + 8];
            bg.q[0]  = *(const uint4*)&sB[cur][NT + n][kb2];
            bg.q[1]  = *(const uint4*)&sB[cur][NT + n][kb2 + 8];
#pragma unroll
            for (int mf = 0; mf < 2; ++mf) {
                acc_kv[mf][nf] = __builtin_amdgcn_wmma_f32_16x16x32_bf16(
                    false, afrag[mf].v, false, bkv.v, (short)0, acc_kv[mf][nf], false, false);
                acc_g[mf][nf] = __builtin_amdgcn_wmma_f32_16x16x32_bf16(
                    false, afrag[mf].v, false, bg.v, (short)0, acc_g[mf][nf], false, false);
            }
        }
        cur ^= 1;
    }

    // ---- epilogue: sigmoid gating, store gated kv ----
    const int rowadd = (lane >> 4) * 8;    // lanes 16..31 hold M=v+8
#pragma unroll
    for (int mf = 0; mf < 2; ++mf) {
#pragma unroll
        for (int nf = 0; nf < 4; ++nf) {
            const int mbase = m0 + wm * 32 + mf * 16 + rowadd;
            const int ncol  = n0 + wn * 64 + nf * 16 + (lane & 15);
#pragma unroll
            for (int v = 0; v < 8; ++v) {
                const float g   = acc_g[mf][nf][v];
                const float val = acc_kv[mf][nf][v] * (1.0f / (1.0f + __expf(-g)));
                kvout[(size_t)(mbase + v) * OUTC + ncol] = val;
            }
        }
    }
}

// ---------------------------------------------------------------------------
// Kernel 3: combine over RATIO with softmax weights, RMSNorm, RoPE.
// ---------------------------------------------------------------------------
__global__ __launch_bounds__(512)
void combine_kernel(const float* __restrict__ kv,
                    const float* __restrict__ w8,
                    const float* __restrict__ w0,
                    const float* __restrict__ cosp,
                    const float* __restrict__ sinp,
                    const float* __restrict__ norm_w,
                    float* __restrict__ out) {
    __shared__ float red[HD];

    const int bc = blockIdx.x;
    const int b  = bc >> 10;
    const int c  = bc & (CHK - 1);
    const int d  = threadIdx.x;

    const float* base = kv + ((size_t)b * SEQ + (size_t)c * RATIO) * OUTC;
    float acc = 0.f;
    if (c == 0) {
#pragma unroll
        for (int r = 0; r < RATIO; ++r)
            acc += base[r * OUTC + d] * w0[r * HD + d];
    } else {
#pragma unroll
        for (int r = 0; r < RATIO; ++r)
            acc += base[r * OUTC + d] * w8[r * HD + d];
        const float* pbase = kv + ((size_t)b * SEQ + (size_t)(c - 1) * RATIO) * OUTC + HD;
#pragma unroll
        for (int r = 0; r < RATIO; ++r)
            acc += pbase[r * OUTC + d] * w8[(RATIO + r) * HD + d];
    }

    red[d] = acc * acc;
    __syncthreads();
#pragma unroll
    for (int s = HD / 2; s > 0; s >>= 1) {
        if (d < s) red[d] += red[d + s];
        __syncthreads();
    }
    const float scale = rsqrtf(red[0] / (float)HD + EPS);
    __syncthreads();

    const float y = acc * scale * norm_w[d];
    red[d] = y;
    __syncthreads();

    float outv = y;
    if (d >= HD - ROPE) {
        const int i = d - (HD - ROPE);
        const int p = i >> 1;
        const float e  = red[(HD - ROPE) + 2 * p];
        const float o  = red[(HD - ROPE) + 2 * p + 1];
        const size_t ci = ((size_t)b * CHK + c) * (ROPE / 2) + p;
        const float cs = cosp[ci];
        const float sn = sinp[ci];
        outv = (i & 1) ? (e * sn + o * cs) : (e * cs - o * sn);
    }
    out[((size_t)b * CHK + c) * HD + d] = outv;
}

// ---------------------------------------------------------------------------
// Launch
// ---------------------------------------------------------------------------
extern "C" void kernel_launch(void* const* d_in, const int* in_sizes, int n_in,
                              void* d_out, int out_size, void* d_ws, size_t ws_size,
                              hipStream_t stream) {
    const float* x      = (const float*)d_in[0];
    const float* cosp   = (const float*)d_in[1];
    const float* sinp   = (const float*)d_in[2];
    const float* W_kv   = (const float*)d_in[3];
    const float* W_gate = (const float*)d_in[4];
    const float* ape    = (const float*)d_in[5];
    const float* norm_w = (const float*)d_in[6];
    float* out = (float*)d_out;

    // ---- workspace layout (bytes) ----
    char* ws = (char*)d_ws;
    const size_t kv_bytes  = (size_t)BATCH * SEQ * OUTC * sizeof(float);     // 64 MB
    float* kvbuf = (float*)ws;                         ws += kv_bytes;
    float* w8    = (float*)ws;                         ws += 8 * HD * sizeof(float);
    float* w0    = (float*)ws;                         ws += 4 * HD * sizeof(float);
    unsigned short* xbf   = (unsigned short*)ws;       ws += (size_t)BATCH * SEQ * HID * 2;  // 128 MB
    unsigned short* wkvbf = (unsigned short*)ws;       ws += (size_t)OUTC * HID * 2;         // 8 MB
    unsigned short* wgbf  = (unsigned short*)ws;       // 8 MB

    const int xN8 = (BATCH * SEQ * HID) / 8;           // 8,388,608
    const int wN8 = (OUTC * HID) / 8;                  // 524,288

    // 0) one-shot f32 -> bf16 conversion (halves GEMM-loop HBM traffic)
    cvt_bf16_kernel<<<(xN8 + 255) / 256, 256, 0, stream>>>(x, xbf, xN8);
    cvt_bf16_kernel<<<(wN8 + 255) / 256, 256, 0, stream>>>(W_kv, wkvbf, wN8);
    cvt_bf16_kernel<<<(wN8 + 255) / 256, 256, 0, stream>>>(W_gate, wgbf, wN8);

    // 1) softmax weights
    softmax_w_kernel<<<2, 256, 0, stream>>>(ape, w8, w0);

    // 2) fused gated dual-GEMM (async-to-LDS + WMMA bf16)
    dim3 ggrid((BATCH * SEQ) / MT, OUTC / NT);         // 128 x 8
    gemm_gated_kernel<<<ggrid, 256, 0, stream>>>(xbf, wkvbf, wgbf, kvbuf);

    // 3) combine + RMSNorm + RoPE
    combine_kernel<<<BATCH * CHK, HD, 0, stream>>>(kvbuf, w8, w0, cosp, sinp,
                                                   norm_w, out);
}
